// RNNModel_39075612459428
// MI455X (gfx1250) — matrix operand order repaired and checked by reading, not compiled, not run on hardware
//
#include <hip/hip_runtime.h>

// ---------------------------------------------------------------------------
// LSTM encoder-decoder for MI455X (gfx1250, wave32, WMMA).
//
// Phase 1: xW = input @ W_ih^T + b_ih   (big parallel GEMM, bf16 WMMA, NT=4)
// Phase 2: 1024 sequential steps: gates = xW[t] + h @ W_hh^T + b_hh
//          (WMMA, NT=2 -> 512 waves/step for latency hiding), then
//          elementwise LSTM update. W_hh (8 MB bf16) stays L2-resident
//          (192 MB L2) so per-step weight traffic never touches HBM.
//
// GEMM inner loop: ping-pong double-buffered fragments, K unrolled by 64,
// so loads for the next K-block are in flight during the current WMMAs and
// no cur<-next register copies are generated.
// ---------------------------------------------------------------------------

typedef __bf16 bf16_t;
typedef __bf16 v16bf __attribute__((ext_vector_type(16)));
typedef __bf16 v8bf  __attribute__((ext_vector_type(8)));
typedef float  v8f   __attribute__((ext_vector_type(8)));

#define T_STEPS 512
#define BATCH   64
#define NINP_K  512
#define NHID_K  1024
#define NG      4096   // 4 * NHID

// ---------------------------------------------------------------------------
__global__ void f32_to_bf16_kernel(const float* __restrict__ src,
                                   bf16_t* __restrict__ dst, int n) {
  int i = blockIdx.x * blockDim.x + threadIdx.x;
  int stride = gridDim.x * blockDim.x;
  for (; i < n; i += stride) dst[i] = (bf16_t)src[i];
}

// A fragment (16x32 bf16, ISA 7.12.2): lane holds row M=lane%16;
// elems 0..15 = K kb+16*(lane/16)+e (contiguous, matches B).
__device__ __forceinline__ v16bf load_a_frag(const bf16_t* __restrict__ arow,
                                             int kb, int half) {
  v8bf lo = *(const v8bf*)(arow + kb + 16 * half);
  v8bf hi = *(const v8bf*)(arow + kb + 16 * half + 8);
  v16bf a;
#pragma unroll
  for (int e = 0; e < 8; ++e) { a[e] = lo[e]; a[e + 8] = hi[e]; }
  return a;
}

template <int NT>
struct Frags {
  v16bf a;
  v16bf b[NT];
};

template <int NT>
__device__ __forceinline__ void load_frags(Frags<NT>& f,
                                           const bf16_t* __restrict__ arow,
                                           const bf16_t* const* wp, int kb,
                                           int half) {
  f.a = load_a_frag(arow, kb, half);
#pragma unroll
  for (int t = 0; t < NT; ++t) f.b[t] = *(const v16bf*)(wp[t] + kb);
}

template <int NT>
__device__ __forceinline__ void do_wmma(v8f (&acc)[NT], const Frags<NT>& f) {
#pragma unroll
  for (int t = 0; t < NT; ++t)
    acc[t] = __builtin_amdgcn_wmma_f32_16x16x32_bf16(
        false, f.a, false, f.b[t], (short)0, acc[t], false, false);
}

// ---------------------------------------------------------------------------
// out[m][n] = sum_k A[m][k] * W[n][k] + bias[n] (+ addMN[m][n])
//   A  : [M x K]  bf16 row-major (activations)
//   W  : [NG x K] bf16 row-major (torch layout => B = W^T; a lane's B
//        fragment is 16 contiguous K values of row n -> one 32B load)
//   out: [M x NG] f32
// Wave tile: 16 rows x (16*NT) cols. Block = 8 waves. grid=(NG/(128*NT),M/16).
// K must be a multiple of 64 (here 512 or 1024).
// ---------------------------------------------------------------------------
template <int NT>
__global__ __launch_bounds__(256) void gemm_wmma_kernel(
    const bf16_t* __restrict__ A, const bf16_t* __restrict__ W,
    const float* __restrict__ addMN,   // nullable: [M x NG] (e.g. xW_t)
    const float* __restrict__ bias,    // [NG]
    float* __restrict__ out, int K) {
  const int lane = threadIdx.x & 31;
  const int wave = threadIdx.x >> 5;
  const int half = lane >> 4;   // 0: lanes 0-15, 1: lanes 16-31
  const int l16  = lane & 15;
  const int m0 = blockIdx.y * 16;
  const int n0 = (blockIdx.x * 8 + wave) * (16 * NT);

  const bf16_t* __restrict__ arow = A + (size_t)(m0 + l16) * K;
  const bf16_t* wp[NT];
#pragma unroll
  for (int t = 0; t < NT; ++t)
    wp[t] = W + (size_t)(n0 + 16 * t + l16) * K + 16 * half;

  v8f acc[NT];
#pragma unroll
  for (int t = 0; t < NT; ++t) acc[t] = v8f{};

  // ---- ping-pong pipelined K loop (K % 64 == 0) ----
  Frags<NT> f0, f1;
  load_frags<NT>(f0, arow, wp, 0, half);
  int kb = 0;
  for (; kb + 64 < K; kb += 64) {
    load_frags<NT>(f1, arow, wp, kb + 32, half);
    if (kb + 96 < K) {  // near-scope prefetch two blocks ahead
#pragma unroll
      for (int t = 0; t < NT; ++t)
        __builtin_prefetch((const void*)(wp[t] + kb + 96), 0, 3);
    }
    do_wmma<NT>(acc, f0);
    load_frags<NT>(f0, arow, wp, kb + 64, half);
    do_wmma<NT>(acc, f1);
  }
  // tail: kb == K - 64
  load_frags<NT>(f1, arow, wp, kb + 32, half);
  do_wmma<NT>(acc, f0);
  do_wmma<NT>(acc, f1);

  // D layout: elem r -> (M = r + 8*(lane/16), N = lane%16)
#pragma unroll
  for (int t = 0; t < NT; ++t) {
    const int n = n0 + 16 * t + l16;
    const float bv = bias[n];
#pragma unroll
    for (int r = 0; r < 8; ++r) {
      const int m = m0 + r + 8 * half;
      const size_t idx = (size_t)m * NG + n;
      float v = acc[t][r] + bv;
      if (addMN) v += addMN[idx];
      out[idx] = v;
    }
  }
}

// ---------------------------------------------------------------------------
// Elementwise LSTM cell update (PyTorch gate order i,f,g,o).
// Re-stages h as bf16 for the next step's WMMA A-matrix.
// ---------------------------------------------------------------------------
__global__ __launch_bounds__(256) void lstm_update_kernel(
    const float* __restrict__ gates,
    const float* __restrict__ mask_t,   // [BATCH] or nullptr (decoder)
    float* __restrict__ h, float* __restrict__ c,
    bf16_t* __restrict__ h_bf,
    float* __restrict__ out_t) {        // [BATCH*NHID] or nullptr
  const int idx = blockIdx.x * blockDim.x + threadIdx.x;
  const int b = idx >> 10;          // / NHID
  const int j = idx & (NHID_K - 1); // % NHID
  const float* __restrict__ g = gates + (size_t)b * NG;
  float gi = g[j];
  float gf = g[NHID_K + j];
  float gg = g[2 * NHID_K + j];
  float go = g[3 * NHID_K + j];
  gi = 1.f / (1.f + __expf(-gi));
  gf = 1.f / (1.f + __expf(-gf));
  go = 1.f / (1.f + __expf(-go));
  gg = tanhf(gg);
  const float cprev = c[idx];
  const float hprev = h[idx];
  float cn = gf * cprev + gi * gg;
  float hn = go * tanhf(cn);
  if (mask_t) {  // encoder: keep old state where mask == 0
    const float m = mask_t[b];
    hn = hn * m + hprev * (1.f - m);
    cn = cn * m + cprev * (1.f - m);
  }
  h[idx] = hn;
  c[idx] = cn;
  h_bf[idx] = (bf16_t)hn;
  if (out_t) out_t[idx] = hn;
}

__global__ void init_state_kernel(float* __restrict__ h, float* __restrict__ c,
                                  bf16_t* __restrict__ h_bf) {
  const int idx = blockIdx.x * blockDim.x + threadIdx.x;
  if (idx < BATCH * NHID_K) {
    h[idx] = 0.f;
    c[idx] = 0.f;
    h_bf[idx] = (bf16_t)0.f;
  }
}

// ---------------------------------------------------------------------------
extern "C" void kernel_launch(void* const* d_in, const int* in_sizes, int n_in,
                              void* d_out, int out_size, void* d_ws,
                              size_t ws_size, hipStream_t stream) {
  const float* input    = (const float*)d_in[0];
  const float* mask     = (const float*)d_in[1];
  const float* W_ih_enc = (const float*)d_in[2];
  const float* W_hh_enc = (const float*)d_in[3];
  const float* b_ih_enc = (const float*)d_in[4];
  const float* b_hh_enc = (const float*)d_in[5];
  const float* W_ih_dec = (const float*)d_in[6];
  const float* W_hh_dec = (const float*)d_in[7];
  const float* b_ih_dec = (const float*)d_in[8];
  const float* b_hh_dec = (const float*)d_in[9];
  float* out = (float*)d_out;
  (void)in_sizes; (void)n_in; (void)out_size; (void)ws_size;

  // Workspace carve-up (all 256B-aligned chunks)
  char* ws = (char*)d_ws;
  size_t off = 0;
  auto carve = [&](size_t bytes) -> char* {
    char* p = ws + off;
    off += (bytes + 255) & ~size_t(255);
    return p;
  };
  const size_t M_IN = (size_t)T_STEPS * BATCH;  // 32768
  bf16_t* in_bf      = (bf16_t*)carve(M_IN * NINP_K * sizeof(bf16_t));
  bf16_t* Wih_enc_bf = (bf16_t*)carve((size_t)NG * NINP_K * sizeof(bf16_t));
  bf16_t* Wih_dec_bf = (bf16_t*)carve((size_t)NG * NINP_K * sizeof(bf16_t));
  bf16_t* Whh_enc_bf = (bf16_t*)carve((size_t)NG * NHID_K * sizeof(bf16_t));
  bf16_t* Whh_dec_bf = (bf16_t*)carve((size_t)NG * NHID_K * sizeof(bf16_t));
  float*  xW_enc     = (float*)carve(M_IN * NG * sizeof(float));
  float*  xW_dec     = (float*)carve(M_IN * NG * sizeof(float));
  float*  gates      = (float*)carve((size_t)BATCH * NG * sizeof(float));
  float*  h          = (float*)carve((size_t)BATCH * NHID_K * sizeof(float));
  float*  c          = (float*)carve((size_t)BATCH * NHID_K * sizeof(float));
  bf16_t* h_bf       = (bf16_t*)carve((size_t)BATCH * NHID_K * sizeof(bf16_t));

  // --- Stage activations/weights to bf16 ---
  f32_to_bf16_kernel<<<2048, 256, 0, stream>>>(input, in_bf,
                                               (int)(M_IN * NINP_K));
  f32_to_bf16_kernel<<<1024, 256, 0, stream>>>(W_ih_enc, Wih_enc_bf,
                                               NG * NINP_K);
  f32_to_bf16_kernel<<<1024, 256, 0, stream>>>(W_ih_dec, Wih_dec_bf,
                                               NG * NINP_K);
  f32_to_bf16_kernel<<<1024, 256, 0, stream>>>(W_hh_enc, Whh_enc_bf,
                                               NG * NHID_K);
  f32_to_bf16_kernel<<<1024, 256, 0, stream>>>(W_hh_dec, Whh_dec_bf,
                                               NG * NHID_K);

  // --- Input projections: xW = input @ W_ih^T + b_ih (NT=4 WMMA GEMMs) ---
  dim3 grid_in(NG / 512, (unsigned)(M_IN / 16));  // (8, 2048)
  gemm_wmma_kernel<4><<<grid_in, 256, 0, stream>>>(in_bf, Wih_enc_bf, nullptr,
                                                   b_ih_enc, xW_enc, NINP_K);
  gemm_wmma_kernel<4><<<grid_in, 256, 0, stream>>>(in_bf, Wih_dec_bf, nullptr,
                                                   b_ih_dec, xW_dec, NINP_K);

  // --- Zero initial state ---
  init_state_kernel<<<(BATCH * NHID_K) / 256, 256, 0, stream>>>(h, c, h_bf);

  dim3 grid_step(NG / 256, BATCH / 16);  // (16, 4): 512 waves per step GEMM
  const int upd_blocks = (BATCH * NHID_K) / 256;

  // --- Encoder scan (masked) ---
  for (int t = 0; t < T_STEPS; ++t) {
    gemm_wmma_kernel<2><<<grid_step, 256, 0, stream>>>(
        h_bf, Whh_enc_bf, xW_enc + (size_t)t * BATCH * NG, b_hh_enc, gates,
        NHID_K);
    lstm_update_kernel<<<upd_blocks, 256, 0, stream>>>(
        gates, mask + (size_t)t * BATCH, h, c, h_bf, nullptr);
  }

  // --- Decoder scan (unmasked, emits hidden states) ---
  for (int t = 0; t < T_STEPS; ++t) {
    gemm_wmma_kernel<2><<<grid_step, 256, 0, stream>>>(
        h_bf, Whh_dec_bf, xW_dec + (size_t)t * BATCH * NG, b_hh_dec, gates,
        NHID_K);
    lstm_update_kernel<<<upd_blocks, 256, 0, stream>>>(
        gates, nullptr, h, c, h_bf, out + (size_t)t * BATCH * NHID_K);
  }
}
